// FBSDE_58256936403490
// MI455X (gfx1250) — matrix operand order repaired and checked by reading, not compiled, not run on hardware
//
#include <hip/hip_runtime.h>

// ============================ types ============================
typedef __attribute__((ext_vector_type(16))) __bf16 v16bf;
typedef __attribute__((ext_vector_type(8)))  float  v8f;

// ======================= problem constants =====================
#define MU_C    0.05f
#define SIGMA_C 0.3f
static constexpr int BATCH = 2048;
static constexpr int NSTEP = 400;
static constexpr int LAG   = 8;
static constexpr int NW    = 51;   // NSTEP/LAG + 1
static constexpr int KIN   = 85;   // SIGCH(84) + time
static constexpr int KPAD  = 96;   // pad K for 32-chunks
static constexpr int HID   = 128;
static constexpr int G4    = 512;  // 4*HID
static constexpr int NBG   = BATCH / 16;  // batch groups of 16 rows
static constexpr int TB    = 3;           // time-steps per FFN block (51 = 17*3)

// ========================= helpers =============================
__device__ __forceinline__ unsigned short f2b(float f){
  union { float f; unsigned u; } v; v.f = f;
  unsigned r = v.u + 0x7FFFu + ((v.u >> 16) & 1u);
  return (unsigned short)(r >> 16);
}
__device__ __forceinline__ float b2f(unsigned short h){
  union { unsigned u; float f; } v; v.u = ((unsigned)h) << 16;
  return v.f;
}
__device__ __forceinline__ float sigm(float x){ return 1.0f / (1.0f + __expf(-x)); }
__device__ __forceinline__ v8f vzero8(){
  v8f z = {0.f,0.f,0.f,0.f,0.f,0.f,0.f,0.f}; return z;
}
__device__ __forceinline__ v8f wmma_bf(v16bf a, v16bf b, v8f c){
  // D = A(16x32 bf16) x B(32x16 bf16) + C(f32)
  return __builtin_amdgcn_wmma_f32_16x16x32_bf16(false, a, false, b, (short)0, c, false, false);
}
// A-fragment coords for element (row m in [0,16), k in [0,32)):
//   lane = (m&15) | ((k>>3)&1)<<4 ; elem = ((k>>4)&1)<<3 | (k&7)

// ============ K0: swizzle W[N][K] (f32) -> B-fragment bf16 ============
// B layout (32x16 bf16): lane = (n & 15) + 16*h ; element i -> k = 8*h + (i&7) + 16*(i>>3)
__global__ void k_swizzleB(const float* __restrict__ src, unsigned short* __restrict__ dst,
                           int K, int nKc, int total){
  int e = blockIdx.x * 256 + threadIdx.x;
  if (e >= total) return;
  int frag = e >> 9, q = e & 511, la = q >> 4, i = q & 15;
  int nt = frag / nKc, kc = frag % nKc;
  int n = nt * 16 + (la & 15);
  int k = kc * 32 + ((la >> 4) * 8) + (i & 7) + ((i >> 3) << 4);
  float v = (k < K) ? src[n * K + k] : 0.0f;
  dst[e] = f2b(v);
}

__global__ void k_addbias(const float* a, const float* b, float* dst, int n){
  int i = blockIdx.x * 256 + threadIdx.x;
  if (i < n) dst[i] = a[i] + b[i];
}

// ================= K1: SDE scan + payoff =================
__global__ void k_sde(const float* __restrict__ ts, const float* __restrict__ x0,
                      const float* __restrict__ dW, float* __restrict__ xs,
                      float* __restrict__ payoff, float* __restrict__ loss0){
  int b = blockIdx.x * 256 + threadIdx.x;
  if (b == 0 && blockIdx.x == 0) loss0[0] = 0.0f;   // zero loss accumulator
  if (b >= BATCH) return;
  float xa = x0[b*2+0], xb = x0[b*2+1];
  xs[(size_t)(b*(NSTEP+1))*2 + 0] = xa;
  xs[(size_t)(b*(NSTEP+1))*2 + 1] = xb;
  float mx = xa;
  for (int t = 0; t < NSTEP; t++){
    float h   = ts[t+1] - ts[t];
    float dwa = dW[(size_t)(b*NSTEP + t)*2 + 0];
    float dwb = dW[(size_t)(b*NSTEP + t)*2 + 1];
    xa = xa + MU_C*xa*h + SIGMA_C*xa*dwa;
    xb = xb + MU_C*xb*h + SIGMA_C*xb*dwb;
    xs[(size_t)(b*(NSTEP+1) + t + 1)*2 + 0] = xa;
    xs[(size_t)(b*(NSTEP+1) + t + 1)*2 + 1] = xb;
    mx = fmaxf(mx, xa);
  }
  payoff[b] = mx - xa;
}

// ===== K2: lead-lag depth-3 signature -> tx in A-FRAGMENT layout + dW window sums =====
// txF layout: [bg = b/16][w][kc(3)][512]  (one 16x32 bf16 A-fragment per (bg,w,kc))
__global__ void k_sig(const float* __restrict__ ts, const float* __restrict__ xs,
                      const float* __restrict__ dW, unsigned short* __restrict__ txF,
                      float* __restrict__ sums){
  int tid = blockIdx.x * 128 + threadIdx.x;
  if (tid >= BATCH * NW) return;
  int b = tid / NW, w = tid % NW;
  float p[9][2];
  if (w == 0){
    p[0][0] = 0.f; p[0][1] = 0.f;
    float a = xs[(size_t)(b*(NSTEP+1))*2 + 0];
    float c = xs[(size_t)(b*(NSTEP+1))*2 + 1];
    #pragma unroll
    for (int j = 1; j < 9; j++){ p[j][0] = a; p[j][1] = c; }
  } else {
    int s0 = (w-1) * LAG;
    #pragma unroll
    for (int j = 0; j < 9; j++){
      p[j][0] = xs[(size_t)(b*(NSTEP+1) + s0 + j)*2 + 0];
      p[j][1] = xs[(size_t)(b*(NSTEP+1) + s0 + j)*2 + 1];
    }
  }
  float a1[4] = {0,0,0,0};
  float a2[16]; float a3[64];
  #pragma unroll
  for (int i = 0; i < 16; i++) a2[i] = 0.f;
  #pragma unroll
  for (int i = 0; i < 64; i++) a3[i] = 0.f;
  // lead-lag increments alternate: even s -> (dq, 0), odd s -> (0, dq)
  for (int s = 0; s < 16; s++){
    int j = s >> 1;
    float qa = p[j+1][0] - p[j][0];
    float qb = p[j+1][1] - p[j][1];
    float d1[4] = {0.f, 0.f, 0.f, 0.f};
    if ((s & 1) == 0){ d1[0] = qa; d1[1] = qb; } else { d1[2] = qa; d1[3] = qb; }
    #pragma unroll
    for (int i = 0; i < 4; i++)
      #pragma unroll
      for (int jj = 0; jj < 4; jj++){
        float a2ij = a2[i*4+jj];
        float hd   = 0.5f * d1[i] * d1[jj];       // d2[i][jj]
        float ahd  = 0.5f * a1[i] * d1[jj];       // a1[i]*d2 factor
        #pragma unroll
        for (int k = 0; k < 4; k++)
          a3[(i*4+jj)*4 + k] += a2ij*d1[k] + ahd*d1[k] + hd*d1[k]*(1.0f/3.0f);
      }
    #pragma unroll
    for (int i = 0; i < 4; i++)
      #pragma unroll
      for (int jj = 0; jj < 4; jj++)
        a2[i*4+jj] += a1[i]*d1[jj] + 0.5f*d1[i]*d1[jj];
    #pragma unroll
    for (int i = 0; i < 4; i++) a1[i] += d1[i];
  }
  // scatter the 96 channels into the A-fragment layout (one-time transform)
  size_t base = ((size_t)((b >> 4) * NW + w)) * 3 * 512;
  const int mlo = b & 15;
  auto put = [&](int k, float v){
    int kc = k >> 5;
    int la = mlo | (((k >> 3) & 1) << 4);
    int ii = (((k >> 4) & 1) << 3) | (k & 7);
    txF[base + (size_t)kc*512 + la*16 + ii] = f2b(v);
  };
  put(0, ts[w * LAG]);
  #pragma unroll
  for (int i = 0; i < 4;  i++) put(1 + i,  a1[i]);
  #pragma unroll
  for (int i = 0; i < 16; i++) put(5 + i,  a2[i]);
  #pragma unroll
  for (int i = 0; i < 64; i++) put(21 + i, a3[i]);
  #pragma unroll
  for (int k = KIN; k < KPAD; k++) put(k, 0.f);

  float s0 = 0.f, s1 = 0.f;
  if (w < NW - 1){
    #pragma unroll
    for (int l = 0; l < LAG; l++){
      s0 += dW[(size_t)(b*NSTEP + w*LAG + l)*2 + 0];
      s1 += dW[(size_t)(b*NSTEP + w*LAG + l)*2 + 1];
    }
  }
  sums[(size_t)tid*2 + 0] = s0;
  sums[(size_t)tid*2 + 1] = s1;
}

// ================= K3: persistent LSTM (WMMA bf16, f32 accum) =================
// Block: 256 thr (8 waves), 32 batch rows (2 fragment row-tiles). Wave w owns
// hidden cols [16w,16w+16) across all 4 gates. Cell state in VGPRs. tx A-frags
// stream straight from global; recurrent h exchanged via LDS fragments and
// copied contiguously to global in fragment layout for the FFN kernel.
__global__ __launch_bounds__(256) void k_lstm(
    const unsigned short* __restrict__ txF,
    const unsigned short* __restrict__ WiS0, const unsigned short* __restrict__ WhS0, const float* __restrict__ bz0,
    const unsigned short* __restrict__ WiS1, const unsigned short* __restrict__ WhS1, const float* __restrict__ bz1,
    unsigned short* __restrict__ hsF)
{
  const int net = blockIdx.y;
  const unsigned short* WiS = net ? WiS1 : WiS0;
  const unsigned short* WhS = net ? WhS1 : WhS0;
  const float* bz = net ? bz1 : bz0;
  unsigned short* hsN = hsF + (size_t)net * (size_t)NBG * NW * 4 * 512;

  const int g0   = blockIdx.x * 2;      // first batch group (16 rows each)
  const int tid  = threadIdx.x;
  const int lane = tid & 31;
  const int wv   = tid >> 5;

  __shared__ __align__(32) unsigned short hA[2*4*512]; // h A-frags (2 rt x 4 kc)

  for (int e = tid; e < 2*4*512; e += 256) hA[e] = 0;  // h_{-1} = 0
  __syncthreads();

  const int jloc = wv*16 + (lane & 15);                // hidden unit owned by this lane
  float bg[4];
  #pragma unroll
  for (int g = 0; g < 4; g++) bg[g] = bz[g*128 + jloc];

  float cst[2][8];
  #pragma unroll
  for (int rt = 0; rt < 2; rt++)
    #pragma unroll
    for (int r = 0; r < 8; r++) cst[rt][r] = 0.f;

  const int kcx = jloc >> 5;
  const int laX = (((jloc >> 3) & 1) << 4);
  const int iiX = (((jloc >> 4) & 1) << 3) | (jloc & 7);

  for (int t = 0; t < NW; t++){
    v8f acc[2][4];
    #pragma unroll
    for (int rt = 0; rt < 2; rt++)
      #pragma unroll
      for (int g = 0; g < 4; g++) acc[rt][g] = vzero8();

    // z += tx_t @ Wi^T  (A-frags direct from global, 32B/lane each)
    #pragma unroll
    for (int kc = 0; kc < 3; kc++){
      v16bf a0 = *(const v16bf*)&txF[((size_t)((g0+0)*NW + t)*3 + kc)*512 + lane*16];
      v16bf a1 = *(const v16bf*)&txF[((size_t)((g0+1)*NW + t)*3 + kc)*512 + lane*16];
      #pragma unroll
      for (int g = 0; g < 4; g++){
        int ct = 8*g + wv;
        v16bf bf = *(const v16bf*)&WiS[((size_t)(ct*3 + kc))*512 + lane*16];
        acc[0][g] = wmma_bf(a0, bf, acc[0][g]);
        acc[1][g] = wmma_bf(a1, bf, acc[1][g]);
      }
    }
    // z += h_{t-1} @ Wh^T  (A-frags from LDS)
    #pragma unroll
    for (int kc = 0; kc < 4; kc++){
      v16bf a0 = *(const v16bf*)&hA[(0*4 + kc)*512 + lane*16];
      v16bf a1 = *(const v16bf*)&hA[(1*4 + kc)*512 + lane*16];
      #pragma unroll
      for (int g = 0; g < 4; g++){
        int ct = 8*g + wv;
        v16bf bf = *(const v16bf*)&WhS[((size_t)(ct*4 + kc))*512 + lane*16];
        acc[0][g] = wmma_bf(a0, bf, acc[0][g]);
        acc[1][g] = wmma_bf(a1, bf, acc[1][g]);
      }
    }
    __syncthreads();  // all waves done reading hA(t-1)

    // gates (pure-register), scatter new h into hA fragments
    #pragma unroll
    for (int rt = 0; rt < 2; rt++){
      #pragma unroll
      for (int r = 0; r < 8; r++){
        float zi = acc[rt][0][r] + bg[0];
        float zf = acc[rt][1][r] + bg[1];
        float zg = acc[rt][2][r] + bg[2];
        float zo = acc[rt][3][r] + bg[3];
        float c  = sigm(zf)*cst[rt][r] + sigm(zi)*tanhf(zg);
        cst[rt][r] = c;
        float hv = sigm(zo)*tanhf(c);
        int m = rt*16 + ((lane >> 4)*8) + r;
        hA[((rt*4 + kcx)*512) + ((m & 15) + laX)*16 + iiX] = f2b(hv);
      }
    }
    __syncthreads();  // hA(t) complete

    // contiguous LDS->global copy of the fragments (2x b128 per thread)
    {
      const uint4* src = (const uint4*)hA;           // 512 uint4 total
      #pragma unroll
      for (int v = 0; v < 2; v++){
        int idx = tid + v*256;                       // uint4 index
        int rt  = idx >> 8;                          // 256 uint4 per row-tile
        int rem = idx & 255;
        uint4* dst = (uint4*)&hsN[((size_t)((g0+rt)*NW + t)*4)*512];
        dst[rem] = src[idx];
      }
    }
    // no barrier needed: next write to hA happens after the next __syncthreads()
  }
}

// ================= K4: FFN (two 128x128 WMMA GEMMs + small head) =================
// Block = 48 rows: batch-group bg (16 rows) x 3 consecutive time steps.
// Input A-frags read directly from global hsF (written by k_lstm).
__global__ __launch_bounds__(256) void k_ffn(
    const unsigned short* __restrict__ hsF,
    const unsigned short* __restrict__ W1S0, const float* __restrict__ b1_0,
    const unsigned short* __restrict__ W2S0, const float* __restrict__ b2_0,
    const float* __restrict__ W3_0, const float* __restrict__ b3_0,
    const unsigned short* __restrict__ W1S1, const float* __restrict__ b1_1,
    const unsigned short* __restrict__ W2S1, const float* __restrict__ b2_1,
    const float* __restrict__ W3_1, const float* __restrict__ b3_1,
    float* __restrict__ Yout, float* __restrict__ Zout)
{
  const int net = blockIdx.y;
  const unsigned short* W1S = net ? W1S1 : W1S0;
  const unsigned short* W2S = net ? W2S1 : W2S0;
  const float* b1 = net ? b1_1 : b1_0;
  const float* b2 = net ? b2_1 : b2_0;
  const float* W3 = net ? W3_1 : W3_0;
  const float* b3 = net ? b3_1 : b3_0;
  const unsigned short* hsN = hsF + (size_t)net * (size_t)NBG * NW * 4 * 512;

  const int bgp  = blockIdx.x / (NW / TB);
  const int t0   = (blockIdx.x % (NW / TB)) * TB;
  const int tid  = threadIdx.x;
  const int lane = tid & 31;
  const int wv   = tid >> 5;

  __shared__ __align__(32) unsigned short bufB[3*4*512]; // y1 A-frags (3 rt x 4 kc)
  __shared__ __align__(32) unsigned short y2p [3*16*128]; // y2 plain [48][128]

  __builtin_prefetch(&W1S[(size_t)wv*4*512 + lane*16], 0, 0);
  __builtin_prefetch(&W2S[(size_t)wv*4*512 + lane*16], 0, 0);

  const int jloc = wv*16 + (lane & 15);
  const int kcx  = jloc >> 5;
  const int laX  = (((jloc >> 3) & 1) << 4);
  const int iiX  = (((jloc >> 4) & 1) << 3) | (jloc & 7);

  // GEMM1: y1 = relu(h @ W1^T + b1); A-frags straight from global
  v8f acc[3];
  #pragma unroll
  for (int rt = 0; rt < 3; rt++) acc[rt] = vzero8();
  #pragma unroll
  for (int kc = 0; kc < 4; kc++){
    v16bf bf = *(const v16bf*)&W1S[((size_t)(wv*4 + kc))*512 + lane*16];
    #pragma unroll
    for (int rt = 0; rt < 3; rt++){
      v16bf a = *(const v16bf*)&hsN[((size_t)(bgp*NW + t0 + rt)*4 + kc)*512 + lane*16];
      acc[rt] = wmma_bf(a, bf, acc[rt]);
    }
  }
  float b1v = b1[jloc];
  #pragma unroll
  for (int rt = 0; rt < 3; rt++)
    #pragma unroll
    for (int r = 0; r < 8; r++){
      float y = fmaxf(acc[rt][r] + b1v, 0.f);
      int m = rt*16 + ((lane >> 4)*8) + r;
      bufB[((rt*4 + kcx)*512) + ((m & 15) + laX)*16 + iiX] = f2b(y);
    }
  __syncthreads();

  // GEMM2: y2 = relu(y1 @ W2^T + b2)
  v8f acc2[3];
  #pragma unroll
  for (int rt = 0; rt < 3; rt++) acc2[rt] = vzero8();
  #pragma unroll
  for (int kc = 0; kc < 4; kc++){
    v16bf bf = *(const v16bf*)&W2S[((size_t)(wv*4 + kc))*512 + lane*16];
    #pragma unroll
    for (int rt = 0; rt < 3; rt++){
      v16bf a = *(const v16bf*)&bufB[((rt*4 + kc)*512) + lane*16];
      acc2[rt] = wmma_bf(a, bf, acc2[rt]);
    }
  }
  __syncthreads();
  float b2v = b2[jloc];
  #pragma unroll
  for (int rt = 0; rt < 3; rt++)
    #pragma unroll
    for (int r = 0; r < 8; r++){
      float y = fmaxf(acc2[rt][r] + b2v, 0.f);
      int m = rt*16 + ((lane >> 4)*8) + r;
      y2p[m*128 + jloc] = f2b(y);
    }
  __syncthreads();

  // head: y3 = y2 @ W3^T + b3  (outd = 1 for f-net, 2 for z-net)
  float* part = (float*)bufB;         // safe: GEMM2 reads done (barrier above)
  int m  = tid >> 2;                  // 0..63 (rows 48..63 idle)
  int qd = tid & 3;
  int outd = net ? 2 : 1;
  for (int o = 0; o < outd; o++){
    if (m < 48){
      float s = 0.f;
      #pragma unroll
      for (int j = qd*32; j < qd*32 + 32; j++)
        s += b2f(y2p[m*128 + j]) * W3[o*128 + j];
      part[m*4 + qd] = s;
    }
    __syncthreads();
    if (qd == 0 && m < 48){
      float tot = part[m*4] + part[m*4+1] + part[m*4+2] + part[m*4+3] + b3[o];
      int b = bgp*16 + (m & 15);
      int t = t0 + (m >> 4);
      size_t r = (size_t)b*NW + t;
      if (net == 0) Yout[r] = tot;
      else          Zout[r*2 + o] = tot;
    }
    __syncthreads();
  }
}

// ================= K5: BSDE loss =================
__global__ void k_loss(const float* __restrict__ ts, const float* __restrict__ Y,
                       const float* __restrict__ Z, const float* __restrict__ sums,
                       const float* __restrict__ payoff, float* __restrict__ loss){
  int b = blockIdx.x * 256 + threadIdx.x;
  float acc = 0.f;
  if (b < BATCH){
    for (int w = 0; w < NW - 1; w++){
      float hh    = ts[(w+1)*LAG] - ts[w*LAG];
      float disc  = expf(-MU_C * hh);
      float target= disc * Y[b*NW + (w+1)];
      float stoch = Z[(size_t)(b*NW + w)*2 + 0]*sums[(size_t)(b*NW + w)*2 + 0]
                  + Z[(size_t)(b*NW + w)*2 + 1]*sums[(size_t)(b*NW + w)*2 + 1];
      float d = (Y[b*NW + w] + stoch) - target;
      acc += d * d;
    }
    float d = Y[b*NW + (NW-1)] - payoff[b];
    acc += d * d;
  }
  __shared__ float red[256];
  red[threadIdx.x] = acc;
  __syncthreads();
  for (int s = 128; s > 0; s >>= 1){
    if (threadIdx.x < s) red[threadIdx.x] += red[threadIdx.x + s];
    __syncthreads();
  }
  if (threadIdx.x == 0) atomicAdd(loss, red[0] * (1.0f / BATCH));
}

// ============================ launch ============================
extern "C" void kernel_launch(void* const* d_in, const int* in_sizes, int n_in,
                              void* d_out, int out_size, void* d_ws, size_t ws_size,
                              hipStream_t stream)
{
  (void)n_in; (void)out_size; (void)ws_size;
  const float* ts = (const float*)d_in[0];
  const float* x0 = (const float*)d_in[1];
  const float* dW = (const float*)d_in[2];
  // params pytree order (dict keys sorted): Wh, Wi, bh, bi, ffn[(W,b)x3]
  struct Net { const float *Wi,*Wh,*bi,*bh,*F0w,*F0b,*F1w,*F1b,*F2w,*F2b; };
  auto parse = [&](int base)->Net{
    Net n;
    if (in_sizes[base] == 4*HID*HID){ n.Wh=(const float*)d_in[base];   n.Wi=(const float*)d_in[base+1]; }
    else                            { n.Wi=(const float*)d_in[base];   n.Wh=(const float*)d_in[base+1]; }
    n.bh =(const float*)d_in[base+2]; n.bi =(const float*)d_in[base+3];
    n.F0w=(const float*)d_in[base+4]; n.F0b=(const float*)d_in[base+5];
    n.F1w=(const float*)d_in[base+6]; n.F1b=(const float*)d_in[base+7];
    n.F2w=(const float*)d_in[base+8]; n.F2b=(const float*)d_in[base+9];
    return n;
  };
  Net nf = parse(3), nz = parse(13);

  // workspace carve
  char* ws = (char*)d_ws; size_t off = 0;
  auto carve = [&](size_t bytes)->void*{
    void* p = ws + off; off += (bytes + 255) & ~(size_t)255; return p;
  };
  float*          xs   = (float*)         carve(sizeof(float)*(size_t)BATCH*(NSTEP+1)*2);
  unsigned short* txF  = (unsigned short*)carve(2ull*NBG*NW*3*512);
  float*          sums = (float*)         carve(sizeof(float)*(size_t)BATCH*NW*2);
  float*          Zb   = (float*)         carve(sizeof(float)*(size_t)BATCH*NW*2);
  unsigned short* hsF  = (unsigned short*)carve(2ull*2*NBG*NW*4*512);
  unsigned short* WiS0 = (unsigned short*)carve(2ull*32*3*512);
  unsigned short* WiS1 = (unsigned short*)carve(2ull*32*3*512);
  unsigned short* WhS0 = (unsigned short*)carve(2ull*32*4*512);
  unsigned short* WhS1 = (unsigned short*)carve(2ull*32*4*512);
  unsigned short* W1S0 = (unsigned short*)carve(2ull*8*4*512);
  unsigned short* W1S1 = (unsigned short*)carve(2ull*8*4*512);
  unsigned short* W2S0 = (unsigned short*)carve(2ull*8*4*512);
  unsigned short* W2S1 = (unsigned short*)carve(2ull*8*4*512);
  float*          bz0  = (float*)         carve(sizeof(float)*G4);
  float*          bz1  = (float*)         carve(sizeof(float)*G4);

  float* out    = (float*)d_out;
  float* Yout   = out + 1;
  float* payoff = out + 1 + BATCH*NW;

  dim3 b256(256);
  // weight prep (B-fragment bf16 swizzles)
  k_swizzleB<<<(32*3*512 + 255)/256, b256, 0, stream>>>(nf.Wi,  WiS0, KIN, 3, 32*3*512);
  k_swizzleB<<<(32*3*512 + 255)/256, b256, 0, stream>>>(nz.Wi,  WiS1, KIN, 3, 32*3*512);
  k_swizzleB<<<(32*4*512 + 255)/256, b256, 0, stream>>>(nf.Wh,  WhS0, HID, 4, 32*4*512);
  k_swizzleB<<<(32*4*512 + 255)/256, b256, 0, stream>>>(nz.Wh,  WhS1, HID, 4, 32*4*512);
  k_swizzleB<<<(8*4*512  + 255)/256, b256, 0, stream>>>(nf.F0w, W1S0, HID, 4, 8*4*512);
  k_swizzleB<<<(8*4*512  + 255)/256, b256, 0, stream>>>(nz.F0w, W1S1, HID, 4, 8*4*512);
  k_swizzleB<<<(8*4*512  + 255)/256, b256, 0, stream>>>(nf.F1w, W2S0, HID, 4, 8*4*512);
  k_swizzleB<<<(8*4*512  + 255)/256, b256, 0, stream>>>(nz.F1w, W2S1, HID, 4, 8*4*512);
  k_addbias<<<2, b256, 0, stream>>>(nf.bi, nf.bh, bz0, G4);
  k_addbias<<<2, b256, 0, stream>>>(nz.bi, nz.bh, bz1, G4);

  // SDE paths + payoff (+ zero loss)
  k_sde<<<BATCH/256, b256, 0, stream>>>(ts, x0, dW, xs, payoff, out);
  // signatures -> tx fragments, window sums
  k_sig<<<(BATCH*NW)/128, dim3(128), 0, stream>>>(ts, xs, dW, txF, sums);
  // both LSTMs (blockIdx.y = net)
  k_lstm<<<dim3(BATCH/32, 2), b256, 0, stream>>>(txF, WiS0, WhS0, bz0,
                                                 WiS1, WhS1, bz1, hsF);
  // FFN heads -> Y (d_out) and Z (ws)
  k_ffn<<<dim3(NBG*(NW/TB), 2), b256, 0, stream>>>(hsF,
      W1S0, nf.F0b, W2S0, nf.F1b, nf.F2w, nf.F2b,
      W1S1, nz.F0b, W2S1, nz.F1b, nz.F2w, nz.F2b,
      Yout, Zb);
  // loss
  k_loss<<<BATCH/256, b256, 0, stream>>>(ts, Yout, Zb, sums, payoff, out);
}
